// SimplifiedLTX2Block_65859028517572
// MI455X (gfx1250) — matrix operand order repaired and checked by reading, not compiled, not run on hardware
//
#include <hip/hip_runtime.h>
#include <stdint.h>

#define DEV static __device__ __forceinline__

typedef __attribute__((ext_vector_type(16))) __bf16 v16bf;
typedef __attribute__((ext_vector_type(8)))  float  v8f;

// ---------- small helpers ----------
DEV uint16_t f2b(float f) { __bf16 h = (__bf16)f; return __builtin_bit_cast(uint16_t, h); }
DEV float    b2f(uint16_t u) { return (float)__builtin_bit_cast(__bf16, u); }
DEV uint32_t lds_off(const void* p) { return (uint32_t)(uintptr_t)p; }  // low 32 bits of generic addr = LDS offset

DEV void store4bf(uint16_t* p, float a, float b, float c, float d) {
  union { uint16_t u[4]; uint2 v; } r;
  r.u[0] = f2b(a); r.u[1] = f2b(b); r.u[2] = f2b(c); r.u[3] = f2b(d);
  *(uint2*)p = r.v;
}

// CDNA5 async global->LDS (tracked by ASYNCcnt). GVS mode: saddr base + 32-bit voff.
DEV void async_b128(uint32_t lds, uint32_t voff, const void* base) {
  asm volatile("global_load_async_to_lds_b128 %0, %1, %2"
               :: "v"(lds), "v"(voff), "s"(base) : "memory");
}
DEV void wait_async0() { asm volatile("s_wait_asynccnt 0" ::: "memory"); }
DEV void wait_async4() { asm volatile("s_wait_asynccnt 4" ::: "memory"); }

DEV v8f wmma_bf16(v16bf a, v16bf b, v8f c) {
  return __builtin_amdgcn_wmma_f32_16x16x32_bf16(false, a, false, b, (short)0, c, false, false);
}

// A fragment (16x32 bf16): lane l holds row m0+l; lane-half selects K 0-7/16-23 vs 8-15/24-31.
DEV v16bf frag_a(const uint16_t* s, int row0, int k0, int stride) {
  int lane = threadIdx.x & 31, l = lane & 15, hi = lane >> 4;
  const uint16_t* p = s + (row0 + l) * stride + k0 + hi * 8;
  union { uint4 q[2]; v16bf v; } r;
  r.q[0] = *(const uint4*)p;
  r.q[1] = *(const uint4*)(p + 16);
  return r.v;
}
// B fragment (32x16 bf16): lane l holds column n0+l (= row of W); lane-half selects K 0-7/16-23 vs 8-15/24-31.
DEV v16bf frag_b(const uint16_t* s, int row0, int k0, int stride) {
  int lane = threadIdx.x & 31, l = lane & 15, hi = lane >> 4;
  const uint16_t* p = s + (row0 + l) * stride + k0 + hi * 8;
  union { uint4 q[2]; v16bf v; } r;
  r.q[0] = *(const uint4*)p;
  r.q[1] = *(const uint4*)(p + 16);
  return r.v;
}

// ---------- elementwise kernels ----------
__global__ __launch_bounds__(256) void f32_to_bf16_k(const float* __restrict__ in,
                                                     uint16_t* __restrict__ out, int n4) {
  int gid = blockIdx.x * 256 + threadIdx.x;
  if (gid >= n4) return;
  size_t e = (size_t)gid * 4;
  float4 x = *(const float4*)(in + e);
  store4bf(out + e, x.x, x.y, x.z, x.w);
}

// norm_hidden(msa) = hs*(1+scale_msa)+shift_msa  -> bf16
__global__ __launch_bounds__(256) void mod_norm_msa(const float* __restrict__ hs,
                                                    const float* __restrict__ temb,
                                                    const float* __restrict__ sst,
                                                    uint16_t* __restrict__ xb) {
  size_t gid = (size_t)blockIdx.x * 256 + threadIdx.x;
  size_t e = gid * 4;
  int col = (int)(e & 2047);
  int row = (int)(e >> 11);
  int b = row >> 11;
  const float* tb = temb + (size_t)b * 2048;
  float4 x = *(const float4*)(hs + e);
  union { float4 v; float a[4]; } u; u.v = x;
  float o[4];
#pragma unroll
  for (int j = 0; j < 4; ++j) {
    int c = col + j;
    float sc = sst[2048 + c] + tb[c];   // scale_msa (idx 1)
    float sh = sst[c] + tb[c];          // shift_msa (idx 0)
    o[j] = u.a[j] * (1.0f + sc) + sh;
  }
  store4bf(xb + e, o[0], o[1], o[2], o[3]);
}

// hs2 = hs + gate_msa*ao ; xb2 = bf16(hs2*(1+scale_mlp)+shift_mlp)
__global__ __launch_bounds__(256) void residual_mlp_norm(const float* __restrict__ hs,
                                                         const float* __restrict__ ao,
                                                         const float* __restrict__ temb,
                                                         const float* __restrict__ sst,
                                                         float* __restrict__ hs2,
                                                         uint16_t* __restrict__ xb2) {
  size_t gid = (size_t)blockIdx.x * 256 + threadIdx.x;
  size_t e = gid * 4;
  int col = (int)(e & 2047);
  int row = (int)(e >> 11);
  int b = row >> 11;
  const float* tb = temb + (size_t)b * 2048;
  union { float4 v; float a[4]; } x, aoo;
  x.v = *(const float4*)(hs + e);
  aoo.v = *(const float4*)(ao + e);
  float h2[4], o[4];
#pragma unroll
  for (int j = 0; j < 4; ++j) {
    int c = col + j;
    float gm = sst[2 * 2048 + c] + tb[c];  // gate_msa (idx 2)
    float sh = sst[3 * 2048 + c] + tb[c];  // shift_mlp (idx 3)
    float sc = sst[4 * 2048 + c] + tb[c];  // scale_mlp (idx 4)
    h2[j] = x.a[j] + gm * aoo.a[j];
    o[j] = h2[j] * (1.0f + sc) + sh;
  }
  union { float a[4]; float4 v; } w; w.a[0]=h2[0]; w.a[1]=h2[1]; w.a[2]=h2[2]; w.a[3]=h2[3];
  *(float4*)(hs2 + e) = w.v;
  store4bf(xb2 + e, o[0], o[1], o[2], o[3]);
}

__global__ __launch_bounds__(256) void final_residual(const float* __restrict__ hs2,
                                                      const float* __restrict__ ffo,
                                                      const float* __restrict__ temb,
                                                      const float* __restrict__ sst,
                                                      float* __restrict__ out) {
  size_t gid = (size_t)blockIdx.x * 256 + threadIdx.x;
  size_t e = gid * 4;
  int col = (int)(e & 2047);
  int row = (int)(e >> 11);
  int b = row >> 11;
  const float* tb = temb + (size_t)b * 2048;
  union { float4 v; float a[4]; } x, f;
  x.v = *(const float4*)(hs2 + e);
  f.v = *(const float4*)(ffo + e);
  union { float a[4]; float4 v; } w;
#pragma unroll
  for (int j = 0; j < 4; ++j) {
    int c = col + j;
    float gm = sst[5 * 2048 + c] + tb[c];  // gate_mlp (idx 5)
    w.a[j] = x.a[j] + gm * f.a[j];
  }
  *(float4*)(out + e) = w.v;
}

// per-head LN over 128 contiguous elements (one wave per segment) -> bf16
__global__ __launch_bounds__(256) void ln_head(const float* __restrict__ X,
                                               uint16_t* __restrict__ Y) {
  int w = (int)(((size_t)blockIdx.x * 256 + threadIdx.x) >> 5);
  int lane = threadIdx.x & 31;
  const float* p = X + (size_t)w * 128 + lane * 4;
  float4 x = *(const float4*)p;
  float s1 = x.x + x.y + x.z + x.w;
  float s2 = x.x * x.x + x.y * x.y + x.z * x.z + x.w * x.w;
#pragma unroll
  for (int off = 16; off; off >>= 1) {
    s1 += __shfl_xor(s1, off, 32);
    s2 += __shfl_xor(s2, off, 32);
  }
  float mu = s1 * (1.0f / 128.0f);
  float var = s2 * (1.0f / 128.0f) - mu * mu;
  float rs = rsqrtf(var + 1e-5f);
  store4bf(Y + (size_t)w * 128 + lane * 4,
           (x.x - mu) * rs, (x.y - mu) * rs, (x.z - mu) * rs, (x.w - mu) * rs);
}

// V [b,s,h*128+d] f32 -> Vt [b*16+h][d][s] bf16 via LDS tile transpose
__global__ __launch_bounds__(256) void v_transpose(const float* __restrict__ V,
                                                   uint16_t* __restrict__ T) {
  __shared__ __align__(16) uint16_t tl[128 * 136];
  const int t = threadIdx.x;
  const int s0 = blockIdx.x * 128, h = blockIdx.y, b = blockIdx.z;
  const size_t base = ((size_t)b * 2048 + s0) * 2048 + h * 128;
#pragma unroll
  for (int i = 0; i < 16; ++i) {
    int c = t + i * 256;
    int sr = c >> 5, d4 = (c & 31) * 4;
    float4 x = *(const float4*)(V + base + (size_t)sr * 2048 + d4);
    uint16_t* q = &tl[sr * 136 + d4];
    q[0] = f2b(x.x); q[1] = f2b(x.y); q[2] = f2b(x.z); q[3] = f2b(x.w);
  }
  __syncthreads();
  const size_t obase = ((size_t)(b * 16 + h)) * 128 * 2048 + s0;
#pragma unroll
  for (int i = 0; i < 8; ++i) {
    int c = t + i * 256;
    int d = c >> 4, sg = (c & 15) * 8;
    union { uint16_t u[8]; uint4 v; } r;
#pragma unroll
    for (int j = 0; j < 8; ++j) r.u[j] = tl[(sg + j) * 136 + d];
    *(uint4*)(T + obase + (size_t)d * 2048 + sg) = r.v;
  }
}

// GEGLU: g = ff1 * gelu_tanh(ff2)
__global__ __launch_bounds__(256) void geglu_k(const uint16_t* __restrict__ ff,
                                               uint16_t* __restrict__ g) {
  size_t gid = (size_t)blockIdx.x * 256 + threadIdx.x;
  size_t e = gid * 4;
  size_t row = e >> 13;
  int c = (int)(e & 8191);
  const uint16_t* pa = ff + row * 16384 + c;
  const uint16_t* pg = pa + 8192;
  float o[4];
#pragma unroll
  for (int j = 0; j < 4; ++j) {
    float a = b2f(pa[j]);
    float xg = b2f(pg[j]);
    float t3 = 0.7978845608028654f * (xg + 0.044715f * xg * xg * xg);
    o[j] = a * (0.5f * xg * (1.0f + tanhf(t3)));
  }
  store4bf(g + e, o[0], o[1], o[2], o[3]);
}

// ---------- bf16 WMMA GEMM: C[M,N] = A[M,K] * W[N,K]^T + bias ----------
__global__ __launch_bounds__(256) void gemm_bf16_nt(const uint16_t* __restrict__ A,
                                                    const uint16_t* __restrict__ W,
                                                    const float* __restrict__ bias,
                                                    float* __restrict__ Cf,
                                                    uint16_t* __restrict__ Cb,
                                                    int M, int N, int K, int bf16_out) {
  constexpr int STR = 40;  // 32 + 8 pad (80B rows, 16B aligned)
  __shared__ __align__(16) uint16_t At[2][128 * STR];
  __shared__ __align__(16) uint16_t Bt[2][128 * STR];
  const int t = threadIdx.x, lane = t & 31, wid = t >> 5;
  const int wm = wid >> 1, wn = wid & 1;
  const int l = lane & 15, hi = lane >> 4;

  // per-thread async-load slots: two 16B chunks per tile per matrix
  uint32_t aElem[2], bElem[2];
  int ldsE[2];
#pragma unroll
  for (int i = 0; i < 2; ++i) {
    int c = t + i * 256;
    int row = c >> 2, part = c & 3;
    aElem[i] = (uint32_t)(blockIdx.y * 128 + row) * (uint32_t)K + part * 8;
    bElem[i] = (uint32_t)(blockIdx.x * 128 + row) * (uint32_t)K + part * 8;
    ldsE[i] = row * STR + part * 8;
  }
  auto issue = [&](int buf, int kt) {
#pragma unroll
    for (int i = 0; i < 2; ++i) {
      async_b128(lds_off(&At[buf][ldsE[i]]), (aElem[i] + (uint32_t)kt * 32u) * 2u, A);
      async_b128(lds_off(&Bt[buf][ldsE[i]]), (bElem[i] + (uint32_t)kt * 32u) * 2u, W);
    }
  };

  v8f acc[2][4] = {};
  const int KT = K >> 5;
  issue(0, 0);
  for (int kt = 0; kt < KT; ++kt) {
    int cur = kt & 1;
    if (kt + 1 < KT) { issue(cur ^ 1, kt + 1); wait_async4(); }
    else             { wait_async0(); }
    __syncthreads();
    const uint16_t* As = &At[cur][0];
    const uint16_t* Bs = &Bt[cur][0];
    v16bf a0 = frag_a(As, wm * 32, 0, STR);
    v16bf a1 = frag_a(As, wm * 32 + 16, 0, STR);
#pragma unroll
    for (int nt = 0; nt < 4; ++nt) {
      v16bf bb = frag_b(Bs, wn * 64 + nt * 16, 0, STR);
      acc[0][nt] = wmma_bf16(a0, bb, acc[0][nt]);
      acc[1][nt] = wmma_bf16(a1, bb, acc[1][nt]);
    }
    __syncthreads();
  }
#pragma unroll
  for (int mt = 0; mt < 2; ++mt)
#pragma unroll
    for (int nt = 0; nt < 4; ++nt) {
      int col = blockIdx.x * 128 + wn * 64 + nt * 16 + l;
      int row = blockIdx.y * 128 + wm * 32 + mt * 16 + hi * 8;
      float bv = bias[col];
#pragma unroll
      for (int r = 0; r < 8; ++r) {
        float v = acc[mt][nt][r] + bv;
        size_t idx = (size_t)(row + r) * N + col;
        if (bf16_out) Cb[idx] = f2b(v);
        else          Cf[idx] = v;
      }
    }
}

// ---------- flash attention (bf16 WMMA, online softmax) ----------
__global__ __launch_bounds__(256) void attn_flash(const uint16_t* __restrict__ Q,
                                                  const uint16_t* __restrict__ Kb,
                                                  const uint16_t* __restrict__ Vt,
                                                  uint16_t* __restrict__ O) {
  constexpr int S = 2048, H = 2048;
  __shared__ __align__(16) uint16_t Qt[64 * 136];
  __shared__ __align__(16) uint16_t Kt[64 * 136];
  __shared__ __align__(16) uint16_t Vl[128 * 72];
  __shared__ __align__(16) uint16_t Pt[64 * 72];
  __shared__ float pmax[2][64], psum[2][64], mrow[64], lrow[64], crow[64];

  const int t = threadIdx.x, lane = t & 31, wid = t >> 5;
  const int wm = wid & 3, wn = wid >> 2;
  const int l = lane & 15, hi = lane >> 4;
  const int q0 = blockIdx.x * 64, h = blockIdx.y, b = blockIdx.z;
  const size_t qBase = ((size_t)b * S + q0) * H + h * 128;
  const size_t kBase = ((size_t)b * S) * H + h * 128;
  const size_t vBase = ((size_t)(b * 16 + h)) * 128 * S;
  const float scale = 0.08838834764831845f;  // 1/sqrt(128)

  if (t < 64) { mrow[t] = -1e30f; lrow[t] = 0.0f; }
  v8f o[4] = {};

  // Q tile (persistent)
#pragma unroll
  for (int i = 0; i < 4; ++i) {
    int c = t + i * 256, row = c >> 4, part = c & 15;
    async_b128(lds_off(&Qt[row * 136 + part * 8]),
               (uint32_t)((qBase + (size_t)row * H + part * 8) * 2), Q);
  }

  for (int kv0 = 0; kv0 < S; kv0 += 64) {
#pragma unroll
    for (int i = 0; i < 2; ++i) {
      int c = t + i * 256, row = c >> 3, part = c & 7;
      async_b128(lds_off(&Kt[row * 136 + part * 8]),
                 (uint32_t)((kBase + (size_t)(kv0 + row) * H + part * 8) * 2), Kb);
    }
#pragma unroll
    for (int i = 0; i < 4; ++i) {
      int c = t + i * 256, d = c >> 3, part = c & 7;
      async_b128(lds_off(&Vl[d * 72 + part * 8]),
                 (uint32_t)((vBase + (size_t)d * S + kv0 + part * 8) * 2), Vt);
    }
    wait_async0();
    __syncthreads();

    // S = Q K^T * scale  (wave: 16 q-rows x 32 kv-cols)
    v8f sa[2] = {};
#pragma unroll
    for (int kc = 0; kc < 4; ++kc) {
      v16bf a = frag_a(Qt, wm * 16, kc * 32, 136);
#pragma unroll
      for (int nt = 0; nt < 2; ++nt) {
        v16bf bb = frag_b(Kt, wn * 32 + nt * 16, kc * 32, 136);
        sa[nt] = wmma_bf16(a, bb, sa[nt]);
      }
    }
    float rm[8];
#pragma unroll
    for (int r = 0; r < 8; ++r) {
      sa[0][r] *= scale; sa[1][r] *= scale;
      rm[r] = fmaxf(sa[0][r], sa[1][r]);
    }
#pragma unroll
    for (int r = 0; r < 8; ++r)
#pragma unroll
      for (int off = 8; off; off >>= 1) rm[r] = fmaxf(rm[r], __shfl_xor(rm[r], off, 32));
    if (l == 0) {
#pragma unroll
      for (int r = 0; r < 8; ++r) pmax[wn][wm * 16 + hi * 8 + r] = rm[r];
    }
    __syncthreads();
    if (t < 64) {
      float tm = fmaxf(pmax[0][t], pmax[1][t]);
      float mo = mrow[t];
      float mn = fmaxf(mo, tm);
      float cc = __expf(mo - mn);
      mrow[t] = mn; crow[t] = cc; lrow[t] *= cc;
    }
    __syncthreads();

    float mr[8], cr[8], rs[8];
#pragma unroll
    for (int r = 0; r < 8; ++r) {
      int rr = wm * 16 + hi * 8 + r;
      mr[r] = mrow[rr]; cr[r] = crow[rr]; rs[r] = 0.0f;
    }
#pragma unroll
    for (int ot = 0; ot < 4; ++ot)
#pragma unroll
      for (int r = 0; r < 8; ++r) o[ot][r] *= cr[r];
#pragma unroll
    for (int nt = 0; nt < 2; ++nt)
#pragma unroll
      for (int r = 0; r < 8; ++r) {
        float p = __expf(sa[nt][r] - mr[r]);
        rs[r] += p;
        Pt[(wm * 16 + hi * 8 + r) * 72 + wn * 32 + nt * 16 + l] = f2b(p);
      }
#pragma unroll
    for (int r = 0; r < 8; ++r)
#pragma unroll
      for (int off = 8; off; off >>= 1) rs[r] += __shfl_xor(rs[r], off, 32);
    if (l == 0) {
#pragma unroll
      for (int r = 0; r < 8; ++r) psum[wn][wm * 16 + hi * 8 + r] = rs[r];
    }
    __syncthreads();
    if (t < 64) lrow[t] += psum[0][t] + psum[1][t];

    // O += P * V   (wave: 16 q-rows x 64 d-cols)
#pragma unroll
    for (int kc = 0; kc < 2; ++kc) {
      v16bf a = frag_a(Pt, wm * 16, kc * 32, 72);
#pragma unroll
      for (int ot = 0; ot < 4; ++ot) {
        v16bf bb = frag_b(Vl, wn * 64 + ot * 16, kc * 32, 72);
        o[ot] = wmma_bf16(a, bb, o[ot]);
      }
    }
    __syncthreads();
  }

  float li[8];
#pragma unroll
  for (int r = 0; r < 8; ++r) li[r] = 1.0f / lrow[wm * 16 + hi * 8 + r];
#pragma unroll
  for (int ot = 0; ot < 4; ++ot)
#pragma unroll
    for (int r = 0; r < 8; ++r) {
      size_t idx = ((size_t)b * S + q0 + wm * 16 + hi * 8 + r) * H
                 + h * 128 + wn * 64 + ot * 16 + l;
      O[idx] = f2b(o[ot][r] * li[r]);
    }
}

// ---------- host-side orchestration ----------
extern "C" void kernel_launch(void* const* d_in, const int* in_sizes, int n_in,
                              void* d_out, int out_size, void* d_ws, size_t ws_size,
                              hipStream_t stream) {
  const float* hs   = (const float*)d_in[0];
  const float* temb = (const float*)d_in[1];
  const float* Wq = (const float*)d_in[2];   const float* bq = (const float*)d_in[3];
  const float* Wk = (const float*)d_in[4];   const float* bk = (const float*)d_in[5];
  const float* Wv = (const float*)d_in[6];   const float* bv = (const float*)d_in[7];
  const float* Wo = (const float*)d_in[8];   const float* bo = (const float*)d_in[9];
  const float* Wff = (const float*)d_in[10]; const float* bff = (const float*)d_in[11];
  const float* Wfo = (const float*)d_in[12]; const float* bfo = (const float*)d_in[13];
  const float* sst = (const float*)d_in[14];
  float* out = (float*)d_out;

  uint8_t* ws = (uint8_t*)d_ws;
  size_t off = 0;
  auto alloc = [&](size_t bytes) -> uint8_t* {
    uint8_t* p = ws + off;
    off += (bytes + 255) & ~(size_t)255;
    return p;
  };
  uint16_t* wq_b = (uint16_t*)alloc(2048ull * 2048 * 2);
  uint16_t* wk_b = (uint16_t*)alloc(2048ull * 2048 * 2);
  uint16_t* wv_b = (uint16_t*)alloc(2048ull * 2048 * 2);
  uint16_t* wo_b = (uint16_t*)alloc(2048ull * 2048 * 2);
  uint16_t* wff_b = (uint16_t*)alloc(16384ull * 2048 * 2);
  uint16_t* wfo_b = (uint16_t*)alloc(2048ull * 8192 * 2);
  uint16_t* xb = (uint16_t*)alloc(4096ull * 2048 * 2);
  uint8_t* R1 = alloc(134217728ull);  // qf|kf|vf|qb|kb  ->  later ffb, then ffo
  uint8_t* R2 = alloc(67108864ull);   // vt|ob|aof       ->  later gb
  float* hs2 = (float*)alloc(4096ull * 2048 * 4);
  uint16_t* xb2 = (uint16_t*)alloc(4096ull * 2048 * 2);

  float* qf = (float*)R1;
  float* kf = (float*)(R1 + 33554432);
  float* vf = (float*)(R1 + 67108864);
  uint16_t* qb = (uint16_t*)(R1 + 100663296);
  uint16_t* kb = (uint16_t*)(R1 + 117440512);
  uint16_t* ffb = (uint16_t*)R1;  // alias (qf..kb dead)
  float* ffo = (float*)R1;        // alias (ffb dead)
  uint16_t* vt = (uint16_t*)R2;
  uint16_t* ob = (uint16_t*)(R2 + 16777216);
  float* aof = (float*)(R2 + 33554432);
  uint16_t* gb = (uint16_t*)R2;   // alias (vt/ob/aof dead)

  dim3 blk(256);
  // weights -> bf16
  f32_to_bf16_k<<<4096, blk, 0, stream>>>(Wq, wq_b, 1048576);
  f32_to_bf16_k<<<4096, blk, 0, stream>>>(Wk, wk_b, 1048576);
  f32_to_bf16_k<<<4096, blk, 0, stream>>>(Wv, wv_b, 1048576);
  f32_to_bf16_k<<<4096, blk, 0, stream>>>(Wo, wo_b, 1048576);
  f32_to_bf16_k<<<32768, blk, 0, stream>>>(Wff, wff_b, 8388608);
  f32_to_bf16_k<<<16384, blk, 0, stream>>>(Wfo, wfo_b, 4194304);

  mod_norm_msa<<<8192, blk, 0, stream>>>(hs, temb, sst, xb);

  dim3 g2048(16, 32);  // (N/128, M/128)
  gemm_bf16_nt<<<g2048, blk, 0, stream>>>(xb, wq_b, bq, qf, nullptr, 4096, 2048, 2048, 0);
  gemm_bf16_nt<<<g2048, blk, 0, stream>>>(xb, wk_b, bk, kf, nullptr, 4096, 2048, 2048, 0);
  gemm_bf16_nt<<<g2048, blk, 0, stream>>>(xb, wv_b, bv, vf, nullptr, 4096, 2048, 2048, 0);

  ln_head<<<8192, blk, 0, stream>>>(qf, qb);
  ln_head<<<8192, blk, 0, stream>>>(kf, kb);
  v_transpose<<<dim3(16, 16, 2), blk, 0, stream>>>(vf, vt);

  attn_flash<<<dim3(32, 16, 2), blk, 0, stream>>>(qb, kb, vt, ob);

  gemm_bf16_nt<<<g2048, blk, 0, stream>>>(ob, wo_b, bo, aof, nullptr, 4096, 2048, 2048, 0);
  residual_mlp_norm<<<8192, blk, 0, stream>>>(hs, aof, temb, sst, hs2, xb2);

  gemm_bf16_nt<<<dim3(128, 32), blk, 0, stream>>>(xb2, wff_b, bff, nullptr, ffb, 4096, 16384, 2048, 1);
  geglu_k<<<32768, blk, 0, stream>>>(ffb, gb);
  gemm_bf16_nt<<<g2048, blk, 0, stream>>>(gb, wfo_b, bfo, ffo, nullptr, 4096, 2048, 8192, 0);

  final_residual<<<8192, blk, 0, stream>>>(hs2, ffo, temb, sst, out);

  (void)in_sizes; (void)n_in; (void)out_size; (void)ws_size;
}
